// hglm_block_37160057045003
// MI455X (gfx1250) — matrix operand order repaired and checked by reading, not compiled, not run on hardware
//
#include <hip/hip_runtime.h>
#include <hip/hip_bf16.h>
#include <math.h>

// ---------------------------------------------------------------------------
// Problem constants (B=32, DIM=256, H=W=64, HEADS=4)
//   gc = 16, n_id = 80, C4 = 64, P = H*W = 4096
// ---------------------------------------------------------------------------

typedef __attribute__((ext_vector_type(16))) __bf16 v16bf;
typedef __attribute__((ext_vector_type(8)))  float  v8f;

union ABu { v16bf v; unsigned u[8]; };

__device__ __forceinline__ unsigned short f2bf(float x) {
    // round-to-nearest-even f32 -> bf16
    unsigned u = __builtin_bit_cast(unsigned, x);
    unsigned r = (u + 0x7FFFu + ((u >> 16) & 1u)) >> 16;
    return (unsigned short)r;
}
__device__ __forceinline__ unsigned packbf(float lo, float hi) {
    return (unsigned)f2bf(lo) | ((unsigned)f2bf(hi) << 16);
}
__device__ __forceinline__ float gelu_exact(float x) {
    return 0.5f * x * (1.0f + erff(x * 0.70710678118654752f));
}

// ---------------------------------------------------------------------------
// Kernel 0a: convert w_qkvl (256x128 f32) to packed bf16 (row stride 64 u32)
// ---------------------------------------------------------------------------
__global__ void k_wconv(const float* __restrict__ w, unsigned short* __restrict__ wbf, int n) {
    int i = blockIdx.x * blockDim.x + threadIdx.x;
    if (i < n) wbf[i] = f2bf(w[i]);
}

// ---------------------------------------------------------------------------
// Kernel 0b: x2 (x channels 128..255) -> bf16, transposed to [b][pixel][ch]
//   (channel contiguous, row = 128 bf16 = 256 B). LDS-tiled so reads are
//   coalesced along pixels and writes coalesced along channels.
//   Block: 256 thr; tile = 32 channels x 128 pixels.
// ---------------------------------------------------------------------------
__global__ void k_xt(const float* __restrict__ x, unsigned short* __restrict__ x2t) {
    __shared__ unsigned short t[32 * 132];        // [ch][px], pad 4 shorts
    int b  = blockIdx.x >> 7;
    int rs = blockIdx.x & 127;
    int cg = rs >> 5;          // channel group 0..3 (32 ch each)
    int pg = rs & 31;          // pixel group 0..31 (128 px each)
    int tid = threadIdx.x;

    #pragma unroll
    for (int it = 0; it < 16; ++it) {             // read 32x128 f32, cvt
        int c = it * 2 + (tid >> 7);              // 0..31
        int p = tid & 127;
        float xv = x[((size_t)b * 256 + 128 + cg * 32 + c) * 4096 + pg * 128 + p];
        t[c * 132 + p] = f2bf(xv);
    }
    __syncthreads();

    unsigned* ou = (unsigned*)x2t;                // row stride 64 u32
    #pragma unroll
    for (int it = 0; it < 8; ++it) {              // write 128 rows x 16 u32
        int p  = it * 16 + (tid >> 4);            // 0..127
        int cu = tid & 15;                        // u32 column within group
        unsigned lo = t[(2 * cu) * 132 + p];
        unsigned hi = t[(2 * cu + 1) * 132 + p];
        ou[((size_t)b * 4096 + pg * 128 + p) * 64 + cg * 16 + cu] = lo | (hi << 16);
    }
}

// ---------------------------------------------------------------------------
// Kernel 1: x1 branch -> output channels [0,128)
// ---------------------------------------------------------------------------
__global__ void k_x1(const float* __restrict__ x,
                     const float* __restrict__ w_hw, const float* __restrict__ b_hw,
                     const float* __restrict__ w_w,  const float* __restrict__ b_w,
                     const float* __restrict__ w_h,  const float* __restrict__ b_h,
                     float* __restrict__ out) {
    int idx = blockIdx.x * blockDim.x + threadIdx.x;   // B*128*4096 threads
    int p  = idx & 4095;
    int ch = (idx >> 12) & 127;
    int b  = idx >> 19;
    int h = p >> 6, w = p & 63;
    const float* xb = x + ((size_t)b * 256 + ch) * 4096;
    float val;
    if (ch < 80) {
        val = xb[p];
    } else if (ch < 96) {
        int g = ch - 80;
        float acc = b_hw[g];
        const float* wk = w_hw + g * 9;
        #pragma unroll
        for (int dy = -1; dy <= 1; ++dy)
            #pragma unroll
            for (int dx = -1; dx <= 1; ++dx) {
                int hh = h + dy, ww = w + dx;
                float xv = (hh >= 0 && hh < 64 && ww >= 0 && ww < 64) ? xb[hh * 64 + ww] : 0.0f;
                acc = fmaf(xv, wk[(dy + 1) * 3 + (dx + 1)], acc);
            }
        val = acc;
    } else if (ch < 112) {
        int g = ch - 96;
        float acc = b_w[g];
        const float* wk = w_w + g * 11;
        #pragma unroll
        for (int t = 0; t < 11; ++t) {
            int ww = w + t - 5;
            float xv = (ww >= 0 && ww < 64) ? xb[h * 64 + ww] : 0.0f;
            acc = fmaf(xv, wk[t], acc);
        }
        val = acc;
    } else {
        int g = ch - 112;
        float acc = b_h[g];
        const float* wk = w_h + g * 11;
        #pragma unroll
        for (int t = 0; t < 11; ++t) {
            int hh = h + t - 5;
            float xv = (hh >= 0 && hh < 64) ? xb[hh * 64 + w] : 0.0f;
            acc = fmaf(xv, wk[t], acc);
        }
        val = acc;
    }
    out[((size_t)b * 256 + ch) * 4096 + p] = gelu_exact(val);
}

// ---------------------------------------------------------------------------
// Kernel 2: main GEMM Y[256,4096] = W[256,128]*X2[128,4096] + b, GELU, route.
//   WG = 256 thr (8 waves) covers 256 ch x 32 px for one batch.
//   Wave w: nt = w&1 (16-px tile), mg = w>>1 (4 M-tiles) -> acc[4] (32 VGPRs,
//   no spills). X tile staged to LDS via GLOBAL_LOAD_ASYNC_TO_LDS_B128
//   (ASYNCcnt), rows padded to 272 B = 17x16 B: async stores stay 16B-aligned
//   and the 32 fragment rows land on distinct banks (17n mod 64 injective).
// ---------------------------------------------------------------------------
__global__ void __launch_bounds__(256) k_qkvl(
        const unsigned short* __restrict__ x2t,
        const unsigned short* __restrict__ wbf,
        const float* __restrict__ bias,
        float* __restrict__ q, float* __restrict__ k,
        float* __restrict__ v, float* __restrict__ out) {
    __shared__ unsigned short xs[32 * 136];   // 8704 B, row stride 272 B

    int b  = blockIdx.x >> 7;      // batch
    int pt = blockIdx.x & 127;     // 32-pixel tile
    int p0 = pt * 32;
    int tid = threadIdx.x;

    // --- async stage: 32 rows x 256 B from x2t into LDS ---
    const char* gbase = (const char*)x2t + ((size_t)b * 4096 + p0) * 256;
    unsigned ldsbase = (unsigned)(size_t)(&xs[0]);
    #pragma unroll
    for (int i = 0; i < 2; ++i) {
        int chunk = tid + i * 256;            // 512 x 16B chunks
        int row = chunk >> 4, seg = chunk & 15;
        const char* ga = gbase + row * 256 + seg * 16;
        unsigned la = ldsbase + row * 272 + seg * 16;
        asm volatile("global_load_async_to_lds_b128 %0, %1, off"
                     :: "v"(la), "v"(ga) : "memory");
    }
    asm volatile("s_wait_asynccnt 0x0" ::: "memory");
    __syncthreads();

    int lane = tid & 31;
    int wv   = tid >> 5;
    int nt   = wv & 1;           // pixel sub-tile (16 px)
    int mg   = wv >> 1;          // M group: tiles mg*4 .. mg*4+3
    int l16  = lane & 15;
    int hl   = lane >> 4;

    v8f acc[4];
    #pragma unroll
    for (int i = 0; i < 4; ++i) acc[i] = v8f{};

    const unsigned* xsu = (const unsigned*)xs;     // row stride 68 u32
    const unsigned* wu  = (const unsigned*)wbf;    // row stride 64 u32

    #pragma unroll
    for (int ks = 0; ks < 4; ++ks) {
        int kb = ks * 32;
        // B fragment: lane (n,hl), elem e -> K = 16*hl + e
        ABu bf;
        const unsigned* xrow = xsu + (nt * 16 + l16) * 68;
        int bb = (kb >> 1) + 8 * hl;
        #pragma unroll
        for (int j = 0; j < 8; ++j) bf.u[j] = xrow[bb + j];

        #pragma unroll
        for (int mi = 0; mi < 4; ++mi) {
            int mt = mg * 4 + mi;
            // A fragment: elem e -> K = 16*hl + e
            ABu af;
            const unsigned* wrow = wu + (mt * 16 + l16) * 64;
            int a0 = (kb >> 1) + 8 * hl;
            #pragma unroll
            for (int j = 0; j < 4; ++j) af.u[j]     = wrow[a0 + j];
            #pragma unroll
            for (int j = 0; j < 4; ++j) af.u[4 + j] = wrow[a0 + 4 + j];

            acc[mi] = __builtin_amdgcn_wmma_f32_16x16x32_bf16(
                false, af.v, false, bf.v, (short)0, acc[mi], false, false);
        }
    }

    // epilogue: bias + GELU, route (o in [mg*64, mg*64+64) -> branch uniform/wave)
    #pragma unroll
    for (int mi = 0; mi < 4; ++mi) {
        #pragma unroll
        for (int r = 0; r < 8; ++r) {
            int o = (mg * 4 + mi) * 16 + r + 8 * hl;   // output channel
            int p = p0 + nt * 16 + l16;                // pixel
            float val = gelu_exact(acc[mi][r] + bias[o]);
            if (o < 64) {
                q[((size_t)b * 64 + o) * 4096 + p] = val;
            } else if (o < 128) {
                k[((size_t)b * 64 + (o - 64)) * 4096 + p] = val;
            } else if (o < 192) {
                v[((size_t)b * 64 + (o - 128)) * 4096 + p] = val;
            } else {
                out[((size_t)b * 256 + 128 + (o - 192)) * 4096 + p] = val;
            }
        }
    }
}

// ---------------------------------------------------------------------------
// Kernel 3: qp = 3x3 sum/9 stride2 pad1, kp = 2x2 max stride2; emit bf16
// ---------------------------------------------------------------------------
__global__ void k_pool(const float* __restrict__ q, const float* __restrict__ k,
                       unsigned short* __restrict__ qfb, unsigned short* __restrict__ kfb) {
    int idx = blockIdx.x * blockDim.x + threadIdx.x;  // B*64*1024 threads
    int j = idx & 31;
    int i = (idx >> 5) & 31;
    int c = (idx >> 10) & 63;
    int b = idx >> 16;
    const float* qb = q + ((size_t)b * 64 + c) * 4096;
    const float* kb = k + ((size_t)b * 64 + c) * 4096;
    float s = 0.0f;
    #pragma unroll
    for (int dy = -1; dy <= 1; ++dy)
        #pragma unroll
        for (int dx = -1; dx <= 1; ++dx) {
            int r = 2 * i + dy, cc = 2 * j + dx;
            if (r >= 0 && r < 64 && cc >= 0 && cc < 64) s += qb[r * 64 + cc];
        }
    float m = kb[(2 * i) * 64 + 2 * j];
    m = fmaxf(m, kb[(2 * i) * 64 + 2 * j + 1]);
    m = fmaxf(m, kb[(2 * i + 1) * 64 + 2 * j]);
    m = fmaxf(m, kb[(2 * i + 1) * 64 + 2 * j + 1]);
    qfb[((size_t)b * 64 + c) * 1024 + i * 32 + j] = f2bf(s * (1.0f / 9.0f));
    kfb[((size_t)b * 64 + c) * 1024 + i * 32 + j] = f2bf(m);
}

// ---------------------------------------------------------------------------
// Kernel 4: qk[b,c,d] = sum_n qf[b,c,n]*kf[b,d,n]  (64x64, K=1024, bf16 in)
//   one WG (512 thr = 16 waves) per batch; wave -> one 16x16 tile
// ---------------------------------------------------------------------------
__global__ void k_qk(const unsigned short* __restrict__ qfb,
                     const unsigned short* __restrict__ kfb,
                     float* __restrict__ qk) {
    int b    = blockIdx.x;
    int lane = threadIdx.x & 31;
    int wv   = threadIdx.x >> 5;   // 0..15
    int mt = wv & 3, nt = wv >> 2;
    int l16 = lane & 15, hl = lane >> 4;
    const unsigned* qrow = (const unsigned*)qfb + ((size_t)b * 64 + mt * 16 + l16) * 512;
    const unsigned* krow = (const unsigned*)kfb + ((size_t)b * 64 + nt * 16 + l16) * 512;
    v8f acc = v8f{};
    for (int ks = 0; ks < 32; ++ks) {
        int kb = ks * 32;
        ABu af, bf;
        int a0 = (kb >> 1) + 8 * hl;
        #pragma unroll
        for (int j = 0; j < 4; ++j) af.u[j]     = qrow[a0 + j];
        #pragma unroll
        for (int j = 0; j < 4; ++j) af.u[4 + j] = qrow[a0 + 4 + j];
        int bb = (kb >> 1) + 8 * hl;
        #pragma unroll
        for (int j = 0; j < 8; ++j) bf.u[j] = krow[bb + j];
        acc = __builtin_amdgcn_wmma_f32_16x16x32_bf16(
            false, af.v, false, bf.v, (short)0, acc, false, false);
    }
    #pragma unroll
    for (int r = 0; r < 8; ++r) {
        int c = mt * 16 + r + 8 * hl;
        int d = nt * 16 + l16;
        qk[(size_t)b * 4096 + c * 64 + d] = acc[r];
    }
}

// ---------------------------------------------------------------------------
// Kernel 5: softmax over axis c (rows) of qk[b, c, d] -> attn
// ---------------------------------------------------------------------------
__global__ void k_softmax(const float* __restrict__ qk, float* __restrict__ attn) {
    int idx = blockIdx.x * blockDim.x + threadIdx.x;   // B*64 threads
    if (idx >= 32 * 64) return;
    int d = idx & 63, b = idx >> 6;
    const float* col = qk + (size_t)b * 4096 + d;
    float m = -INFINITY;
    for (int c = 0; c < 64; ++c) m = fmaxf(m, col[c * 64]);
    float s = 0.0f;
    for (int c = 0; c < 64; ++c) s += expf(col[c * 64] - m);
    float inv = 1.0f / s;
    float* acol = attn + (size_t)b * 4096 + d;
    for (int c = 0; c < 64; ++c) acol[c * 64] = expf(col[c * 64] - m) * inv;
}

// ---------------------------------------------------------------------------
// Kernel 6: out[b,d,m] = sum_c attn[b,c,d]*v[b,c,m] -> out channels [192,256)
//   WG = 256 thr (8 waves): 4 d-tiles x 2 pixel-tiles; K = 64 (2 WMMA steps)
// ---------------------------------------------------------------------------
__global__ void k_out(const float* __restrict__ attn, const float* __restrict__ v,
                      float* __restrict__ out) {
    int b  = blockIdx.x >> 7;        // batch
    int pt = blockIdx.x & 127;       // 32-pixel tile
    int lane = threadIdx.x & 31;
    int wv   = threadIdx.x >> 5;
    int mt = wv & 3, nt = wv >> 2;   // mt: d-tile, nt: 0..1 pixel sub-tile
    int l16 = lane & 15, hl = lane >> 4;
    int px = pt * 32 + nt * 16 + l16;
    const float* ab = attn + (size_t)b * 4096;        // [c][d]
    const float* vb = v + (size_t)b * 64 * 4096;      // [c][m]
    v8f acc = v8f{};
    #pragma unroll
    for (int ks = 0; ks < 2; ++ks) {
        int kb = ks * 32;
        ABu af, bf;
        int d = mt * 16 + l16;
        #pragma unroll
        for (int j = 0; j < 4; ++j) {
            int c0 = kb + 16 * hl + 2 * j;
            af.u[j] = packbf(ab[c0 * 64 + d], ab[(c0 + 1) * 64 + d]);
            int c1 = kb + 16 * hl + 8 + 2 * j;
            af.u[4 + j] = packbf(ab[c1 * 64 + d], ab[(c1 + 1) * 64 + d]);
        }
        #pragma unroll
        for (int j = 0; j < 8; ++j) {
            int c0 = kb + 16 * hl + 2 * j;
            bf.u[j] = packbf(vb[(size_t)c0 * 4096 + px], vb[(size_t)(c0 + 1) * 4096 + px]);
        }
        acc = __builtin_amdgcn_wmma_f32_16x16x32_bf16(
            false, af.v, false, bf.v, (short)0, acc, false, false);
    }
    #pragma unroll
    for (int r = 0; r < 8; ++r) {
        int d = mt * 16 + r + 8 * hl;
        out[((size_t)b * 256 + 192 + d) * 4096 + px] = acc[r];
    }
}

// ---------------------------------------------------------------------------
// Launch
// ---------------------------------------------------------------------------
extern "C" void kernel_launch(void* const* d_in, const int* in_sizes, int n_in,
                              void* d_out, int out_size, void* d_ws, size_t ws_size,
                              hipStream_t stream) {
    const float* x      = (const float*)d_in[0];
    const float* w_hw   = (const float*)d_in[1];
    const float* b_hw   = (const float*)d_in[2];
    const float* w_w    = (const float*)d_in[3];
    const float* b_w    = (const float*)d_in[4];
    const float* w_h    = (const float*)d_in[5];
    const float* b_h    = (const float*)d_in[6];
    const float* w_qkvl = (const float*)d_in[7];
    const float* b_qkvl = (const float*)d_in[8];
    float* out = (float*)d_out;

    // workspace layout (bytes):
    //   wbf 64K | x2t 32M | q 32M | k 32M | v 32M | qfb 4M | kfb 4M
    //   | qk 512K | attn 512K     (~137 MB total)
    char* ws = (char*)d_ws;
    unsigned short* wbf = (unsigned short*)ws;
    unsigned short* x2t = (unsigned short*)(ws + 65536);
    float* q    = (float*)(ws + 65536 + ((size_t)32 << 20));
    float* kk   = q   + (size_t)32 * 64 * 4096;
    float* v    = kk  + (size_t)32 * 64 * 4096;
    unsigned short* qfb = (unsigned short*)(v + (size_t)32 * 64 * 4096);
    unsigned short* kfb = qfb + (size_t)32 * 64 * 1024;
    float* qkm  = (float*)(kfb + (size_t)32 * 64 * 1024);
    float* attn = qkm + (size_t)32 * 64 * 64;

    k_wconv  <<<128, 256, 0, stream>>>(w_qkvl, wbf, 256 * 128);
    k_xt     <<<32 * 4 * 32, 256, 0, stream>>>(x, x2t);
    k_x1     <<<(32 * 128 * 4096) / 256, 256, 0, stream>>>(x, w_hw, b_hw, w_w, b_w, w_h, b_h, out);
    k_qkvl   <<<32 * 128, 256, 0, stream>>>(x2t, wbf, b_qkvl, q, kk, v, out);
    k_pool   <<<(32 * 64 * 1024) / 256, 256, 0, stream>>>(q, kk, qfb, kfb);
    k_qk     <<<32, 512, 0, stream>>>(qfb, kfb, qkm);
    k_softmax<<<(32 * 64 + 255) / 256, 256, 0, stream>>>(qkm, attn);
    k_out    <<<32 * 128, 256, 0, stream>>>(attn, v, out);
}
